// DeltaMPredictor_62646392980249
// MI455X (gfx1250) — matrix operand order repaired and checked by reading, not compiled, not run on hardware
//
#include <hip/hip_runtime.h>

typedef __attribute__((ext_vector_type(16))) _Float16 v16h;
typedef __attribute__((ext_vector_type(8)))  _Float16 v8h;
typedef __attribute__((ext_vector_type(4)))  _Float16 v4h;
typedef __attribute__((ext_vector_type(8)))  float    v8f;
typedef __attribute__((ext_vector_type(4)))  float    v4f;
typedef __attribute__((ext_vector_type(4)))  unsigned int v4u;
typedef __attribute__((ext_vector_type(8)))  int      v8i;
typedef __attribute__((ext_vector_type(4)))  int      v4i;

#define CAT16(lo, hi) __builtin_shufflevector((lo), (hi), 0,1,2,3,4,5,6,7,8,9,10,11,12,13,14,15)

#define B_DIM   16384
#define C_CAM   4
#define D_BB    2048
#define D_HID   512
#define M_ROWS  (B_DIM * C_CAM)   // 65536

static __device__ __forceinline__ v4h cvt4(v4f v) {
  v4h h;
  h.x = (_Float16)v.x; h.y = (_Float16)v.y;
  h.z = (_Float16)v.z; h.w = (_Float16)v.w;
  return h;
}

// ---------------------------------------------------------------------------
// Tensor Data Mover: 2D tile load (f16 elements) global -> LDS.
// D# packed per CDNA5 ISA.
//   pad codes: interval bytes = 8 << interval_code ; pad bytes = 4*(amount_code+1)
// This toolchain exposes the 6-arg builtin (g0, g1, g2, g3, g4, cpol).
// ---------------------------------------------------------------------------
static __device__ __forceinline__ void tdm_load_2d(unsigned lds_addr,
                                                   const _Float16* gptr,
                                                   unsigned tile_w, unsigned tile_h,
                                                   unsigned long long row_stride_elems,
                                                   unsigned pad_en,
                                                   unsigned pad_interval_code,
                                                   unsigned pad_amount_code)
{
  const unsigned long long ga = (unsigned long long)(uintptr_t)gptr;
  v4u g0;
  g0.x = 1u;                                     // count=1, user descriptor
  g0.y = lds_addr;                               // LDS byte address
  g0.z = (unsigned)ga;                           // global_addr[31:0]  (bits 95:64)
  g0.w = ((unsigned)(ga >> 32) & 0x01FFFFFFu)    // global_addr[56:32] (bits 120:96)
         | (2u << 30);                           // type = 2 ("image")
  v8i g1;
  g1[0] = (int)((1u << 16)                       // data_size = 1 -> 2 bytes
                | (pad_en << 20)
                | (pad_interval_code << 22)
                | (pad_amount_code << 25));
  g1[1] = (int)(tile_w << 16);                                  // tensor_dim0[15:0]
  g1[2] = (int)((tile_w >> 16) | (tile_h << 16));               // tensor_dim0 hi | tensor_dim1 lo
  g1[3] = (int)((tile_h >> 16) | (tile_w << 16));               // tensor_dim1 hi | tile_dim0
  g1[4] = (int)(tile_h & 0xFFFFu);                              // tile_dim1 | tile_dim2=0
  g1[5] = (int)(unsigned)row_stride_elems;                      // tensor_dim0_stride lo
  g1[6] = (int)((unsigned)(row_stride_elems >> 32) & 0xFFFFu);  // stride hi | dim1_stride=0
  g1[7] = 0;
  const v4i z4 = {0, 0, 0, 0};
  const v8i z8 = {0, 0, 0, 0, 0, 0, 0, 0};
  __builtin_amdgcn_tensor_load_to_lds(g0, g1, z4, z4, z8, 0);
}

// 6x6 matmul, fully unrolled -> register resident
static __device__ __forceinline__ void mm6(const float* __restrict__ A,
                                           const float* __restrict__ Bm,
                                           float* __restrict__ O) {
#pragma unroll
  for (int i = 0; i < 6; ++i)
#pragma unroll
    for (int j = 0; j < 6; ++j) {
      float s = 0.0f;
#pragma unroll
      for (int k = 0; k < 6; ++k) s = fmaf(A[i * 6 + k], Bm[k * 6 + j], s);
      O[i * 6 + j] = s;
    }
}

// ---------------------------------------------------------------------------
// Kernel 0: one-shot fp32 -> f16 weight conversion (proj_w, w_gate, w_val,
// w_out padded 36->48 rows) so the GEMM kernels can stage via TDM directly.
// ---------------------------------------------------------------------------
__launch_bounds__(256)
__global__ void k_cvt_weights(const float* __restrict__ pw, const float* __restrict__ wg,
                              const float* __restrict__ wv, const float* __restrict__ wo,
                              _Float16* __restrict__ pwh, _Float16* __restrict__ wgh,
                              _Float16* __restrict__ wvh, _Float16* __restrict__ woh)
{
  const int gid = blockIdx.x * 256 + threadIdx.x;
  const int stride = gridDim.x * 256;
  for (int i = gid; i < (512 * 2048) / 4; i += stride)
    *(v4h*)&pwh[i * 4] = cvt4(*(const v4f*)&pw[i * 4]);
  for (int i = gid; i < (4 * 512 * 512) / 4; i += stride) {
    *(v4h*)&wgh[i * 4] = cvt4(*(const v4f*)&wg[i * 4]);
    *(v4h*)&wvh[i * 4] = cvt4(*(const v4f*)&wv[i * 4]);
  }
  for (int i = gid; i < (4 * 48 * 512) / 4; i += stride) {
    const int j = i * 4;
    const int c = j / (48 * 512);
    const int r = j % (48 * 512);
    const int n = r >> 9, k = r & 511;
    v4h h;
    if (n < 36) h = cvt4(*(const v4f*)&wo[((size_t)c * 36 + n) * 512 + k]);
    else { h.x = h.y = h.z = h.w = (_Float16)0.0f; }
    *(v4h*)&woh[j] = h;
  }
}

// ---------------------------------------------------------------------------
// Kernel 1: projection GEMM (K=2048) + bias + per-camera LayerNorm -> xn (f16)
// B tiles staged by TDM (double buffered); A tile converted f32->f16 by VALU.
// ---------------------------------------------------------------------------
__launch_bounds__(256)
__global__ void k_proj_ln(const float* __restrict__ thumb,   // [65536, 2048] f32
                          const _Float16* __restrict__ pwh,  // [512, 2048] f16
                          const float* __restrict__ pb,      // [512]
                          const float* __restrict__ gamma,   // [4, 512]
                          const float* __restrict__ beta,    // [4, 512]
                          _Float16* __restrict__ xn)         // [65536, 512]
{
  __shared__ __attribute__((aligned(16))) _Float16 sA[32 * 40];        // [32][40]
  __shared__ __attribute__((aligned(16))) _Float16 sB[2 * 512 * 40];   // double buffered [512][40]
  __shared__ __attribute__((aligned(16))) float    sOut[32 * 516];
  __shared__ __attribute__((aligned(16))) float    sBias[512];
  __shared__ float sRed[32 * 8 * 2];
  __shared__ float sMu[32], sRs[32];

  const int t    = threadIdx.x;
  const int wave = t >> 5, lane = t & 31;
  const int lm   = lane & 15, lh = lane >> 4;
  const int row0 = blockIdx.x * 32;
  const int m_sub  = (wave & 1) * 16;
  const int n_base = (wave >> 1) * 128;
  const unsigned ldsB = (unsigned)(uintptr_t)&sB[0];

  if (t < 128) *(v4f*)&sBias[t * 4] = *(const v4f*)&pb[t * 4];

  // TDM: stage first B slice [512 rows x 32 f16], row stride 2048, LDS stride 40 f16
  if (wave == 0)
    tdm_load_2d(ldsB, pwh, 32, 512, 2048ull, 1u, 3u, 3u);

  v8f acc[8] = {};

  for (int kk = 0; kk < D_BB / 32; ++kk) {
    const int k0 = kk * 32;
    if (wave == 0 && kk + 1 < D_BB / 32)
      tdm_load_2d(ldsB + (unsigned)(((kk + 1) & 1) * 512 * 40 * 2),
                  pwh + (kk + 1) * 32, 32, 512, 2048ull, 1u, 3u, 3u);
    // stage A: 32 rows x 32 K (one float4 per thread, f32 -> f16)
    {
      const int m = t >> 3, kc = (t & 7) * 4;
      v4f v = *(const v4f*)&thumb[(size_t)(row0 + m) * D_BB + k0 + kc];
      *(v4h*)&sA[m * 40 + kc] = cvt4(v);
    }
    if (kk + 1 < D_BB / 32)
      __builtin_prefetch(&thumb[(size_t)(row0 + (t & 31)) * D_BB + k0 + 32], 0, 1);
    if (wave == 0) {
      if (kk + 1 < D_BB / 32) __builtin_amdgcn_s_wait_tensorcnt(1);
      else                    __builtin_amdgcn_s_wait_tensorcnt(0);
    }
    __syncthreads();

    const _Float16* bb = &sB[(kk & 1) * 512 * 40];
    v8h alo = *(const v8h*)&sA[(m_sub + lm) * 40 + lh * 16];
    v8h ahi = *(const v8h*)&sA[(m_sub + lm) * 40 + lh * 16 + 8];
    v16h af = CAT16(alo, ahi);
    // software-pipelined B-fragment loads (distinct regs -> overlap ds with wmma)
    v8h blo = *(const v8h*)&bb[(n_base + lm) * 40 + lh * 16];
    v8h bhi = *(const v8h*)&bb[(n_base + lm) * 40 + lh * 16 + 8];
#pragma unroll
    for (int nt = 0; nt < 8; ++nt) {
      v8h nlo = blo, nhi = bhi;
      if (nt < 7) {
        const int n = n_base + (nt + 1) * 16 + lm;
        nlo = *(const v8h*)&bb[n * 40 + lh * 16];
        nhi = *(const v8h*)&bb[n * 40 + lh * 16 + 8];
      }
      acc[nt] = __builtin_amdgcn_wmma_f32_16x16x32_f16(false, af, false, CAT16(blo, bhi),
                                                       (short)0, acc[nt], false, false);
      blo = nlo; bhi = nhi;
    }
    __syncthreads();
  }

  // accumulators (+bias) -> LDS for LayerNorm
#pragma unroll
  for (int nt = 0; nt < 8; ++nt) {
    const int n = n_base + nt * 16 + lm;
    const float bia = sBias[n];
#pragma unroll
    for (int r = 0; r < 8; ++r)
      sOut[(m_sub + lh * 8 + r) * 516 + n] = acc[nt][r] + bia;
  }
  __syncthreads();

  {
    const int m = t >> 3, seg = t & 7;
    float s = 0.0f, s2 = 0.0f;
#pragma unroll
    for (int j = 0; j < 64; ++j) {
      const float v = sOut[m * 516 + seg + 8 * j];
      s += v; s2 = fmaf(v, v, s2);
    }
    sRed[(m * 8 + seg) * 2 + 0] = s;
    sRed[(m * 8 + seg) * 2 + 1] = s2;
  }
  __syncthreads();
  if (t < 32) {
    float s = 0.0f, s2 = 0.0f;
#pragma unroll
    for (int j = 0; j < 8; ++j) { s += sRed[(t * 8 + j) * 2]; s2 += sRed[(t * 8 + j) * 2 + 1]; }
    const float mu  = s * (1.0f / 512.0f);
    const float var = s2 * (1.0f / 512.0f) - mu * mu;
    sMu[t] = mu;
    sRs[t] = rsqrtf(var + 1e-5f);
  }
  __syncthreads();

  {
    const int m = t >> 3, seg = t & 7;
    const int g = row0 + m;
    const int cam = g & 3;
    const float mu = sMu[m], rs = sRs[m];
    const float* gp = &gamma[cam * 512 + seg * 64];
    const float* bp = &beta[cam * 512 + seg * 64];
    const float* rp = &sOut[m * 516 + seg * 64];
    _Float16* op = &xn[(size_t)g * 512 + seg * 64];
#pragma unroll
    for (int j8 = 0; j8 < 8; ++j8) {
      v8h h;
#pragma unroll
      for (int j = 0; j < 8; ++j) {
        const float v = (rp[j8 * 8 + j] - mu) * rs * gp[j8 * 8 + j] + bp[j8 * 8 + j];
        h[j] = (_Float16)v;
      }
      *(v8h*)&op[j8 * 8] = h;
    }
  }
}

// ---------------------------------------------------------------------------
// Kernel 2: per-camera SwiGLU + head GEMM + skew + clip + 6x6 expm.
// Activation tile, weight slices (double buffered) and w_out all staged by TDM.
// ---------------------------------------------------------------------------
__launch_bounds__(256)
__global__ void k_swiglu_head_expm(const _Float16* __restrict__ xn,    // [65536, 512]
                                   const _Float16* __restrict__ wgh,   // [4, 512, 512] f16
                                   const _Float16* __restrict__ wvh,   // [4, 512, 512] f16
                                   const _Float16* __restrict__ woh,   // [4, 48, 512] f16 (padded)
                                   float* __restrict__ out)            // [4, 16384, 36]
{
  __shared__ __attribute__((aligned(16))) _Float16 sA[32 * 520];       // activation tile
  __shared__ __attribute__((aligned(16))) _Float16 sStage[2 * 512 * 40]; // weight slices / w_out
  __shared__ __attribute__((aligned(16))) _Float16 sH[32 * 520];
  __shared__ __attribute__((aligned(16))) float    sO[32 * 52];

  const int t    = threadIdx.x;
  const int wave = t >> 5, lane = t & 31;
  const int lm   = lane & 15, lh = lane >> 4;
  const int c    = blockIdx.y;
  const int b0   = blockIdx.x * 32;
  const int m_sub  = (wave & 1) * 16;
  const int n_base = (wave >> 1) * 128;
  const unsigned ldsA = (unsigned)(uintptr_t)&sA[0];
  const unsigned ldsS = (unsigned)(uintptr_t)&sStage[0];

  const _Float16* wgc = wgh + (size_t)c * 512 * 512;
  const _Float16* wvc = wvh + (size_t)c * 512 * 512;

  // TDM: activation tile [32 rows x 512 f16], rows b*4+c (stride 2048 elems),
  // LDS row stride 520 f16 via pad (1024B interval, 16B pad)
  if (wave == 0) {
    tdm_load_2d(ldsA, xn + ((size_t)(b0 * 4 + c)) * 512, 512, 32, 2048ull, 1u, 7u, 3u);
    tdm_load_2d(ldsS, wgc, 32, 512, 512ull, 1u, 3u, 3u);   // gate slice 0
  }

  v8f gsa[8];

#pragma unroll 1
  for (int phase = 0; phase < 2; ++phase) {
    const _Float16* w = (phase == 0) ? wgc : wvc;
    v8f acc[8] = {};
    for (int kk = 0; kk < 16; ++kk) {
      const int k0 = kk * 32;
      if (wave == 0) {
        if (kk + 1 < 16) {
          tdm_load_2d(ldsS + (unsigned)(((kk + 1) & 1) * 512 * 40 * 2),
                      w + (kk + 1) * 32, 32, 512, 512ull, 1u, 3u, 3u);
          __builtin_amdgcn_s_wait_tensorcnt(1);
        } else if (phase == 0) {
          tdm_load_2d(ldsS, wvc, 32, 512, 512ull, 1u, 3u, 3u);  // val slice 0 into buf 0
          __builtin_amdgcn_s_wait_tensorcnt(1);
        } else {
          __builtin_amdgcn_s_wait_tensorcnt(0);
        }
      }
      __syncthreads();

      const _Float16* bb = &sStage[(kk & 1) * 512 * 40];
      v8h alo = *(const v8h*)&sA[(m_sub + lm) * 520 + k0 + lh * 16];
      v8h ahi = *(const v8h*)&sA[(m_sub + lm) * 520 + k0 + lh * 16 + 8];
      v16h af = CAT16(alo, ahi);
      v8h blo = *(const v8h*)&bb[(n_base + lm) * 40 + lh * 16];
      v8h bhi = *(const v8h*)&bb[(n_base + lm) * 40 + lh * 16 + 8];
#pragma unroll
      for (int nt = 0; nt < 8; ++nt) {
        v8h nlo = blo, nhi = bhi;
        if (nt < 7) {
          const int n = n_base + (nt + 1) * 16 + lm;
          nlo = *(const v8h*)&bb[n * 40 + lh * 16];
          nhi = *(const v8h*)&bb[n * 40 + lh * 16 + 8];
        }
        acc[nt] = __builtin_amdgcn_wmma_f32_16x16x32_f16(false, af, false, CAT16(blo, bhi),
                                                         (short)0, acc[nt], false, false);
        blo = nlo; bhi = nhi;
      }
      __syncthreads();
    }
    if (phase == 0) {
#pragma unroll
      for (int nt = 0; nt < 8; ++nt)
#pragma unroll
        for (int r = 0; r < 8; ++r) {
          const float x = acc[nt][r];
          gsa[nt][r] = x / (1.0f + __expf(-x));
        }
    } else {
#pragma unroll
      for (int nt = 0; nt < 8; ++nt) {
        const int n = n_base + nt * 16 + lm;
#pragma unroll
        for (int r = 0; r < 8; ++r)
          sH[(m_sub + lh * 8 + r) * 520 + n] = (_Float16)(gsa[nt][r] * acc[nt][r]);
      }
    }
  }

  // TDM: w_out tile [48 rows x 512 f16] -> sStage with 520 stride
  if (wave == 0) {
    tdm_load_2d(ldsS, woh + (size_t)c * 48 * 512, 512, 48, 512ull, 1u, 7u, 3u);
    __builtin_amdgcn_s_wait_tensorcnt(0);
  }
  __syncthreads();

  // head GEMM: 2 M-tiles x 3 N-tiles across waves 0..5
  if (wave < 6) {
    const int ms = (wave & 1) * 16, n0 = (wave >> 1) * 16;
    v8f oacc = {};
    for (int kk = 0; kk < 16; ++kk) {
      const int k0 = kk * 32;
      v8h alo = *(const v8h*)&sH[(ms + lm) * 520 + k0 + lh * 16];
      v8h ahi = *(const v8h*)&sH[(ms + lm) * 520 + k0 + lh * 16 + 8];
      v8h blo = *(const v8h*)&sStage[(n0 + lm) * 520 + k0 + lh * 16];
      v8h bhi = *(const v8h*)&sStage[(n0 + lm) * 520 + k0 + lh * 16 + 8];
      oacc = __builtin_amdgcn_wmma_f32_16x16x32_f16(false, CAT16(alo, ahi),
                                                    false, CAT16(blo, bhi),
                                                    (short)0, oacc, false, false);
    }
#pragma unroll
    for (int r = 0; r < 8; ++r)
      sO[(ms + lh * 8 + r) * 52 + n0 + lm] = oacc[r];
  }
  __syncthreads();

  if (t < 32) {
    float S[36];
#pragma unroll
    for (int i = 0; i < 6; ++i)
#pragma unroll
      for (int j = 0; j < 6; ++j)
        S[i * 6 + j] = sO[t * 52 + i * 6 + j] - sO[t * 52 + j * 6 + i];

    float fr = 0.0f;
#pragma unroll
    for (int i = 0; i < 36; ++i) fr = fmaf(S[i], S[i], fr);
    fr = sqrtf(fr);
    const float scale = fminf(fr, 3.0f) / fmaxf(fr, 1e-8f);

    float As[36], E[36], term[36], tmp[36];
#pragma unroll
    for (int i = 0; i < 36; ++i) As[i] = S[i] * (scale * 0.0625f);
#pragma unroll
    for (int i = 0; i < 36; ++i) { const float id = (i % 7 == 0) ? 1.0f : 0.0f; E[i] = id; term[i] = id; }
#pragma unroll
    for (int k = 1; k <= 12; ++k) {
      mm6(term, As, tmp);
      const float inv = 1.0f / (float)k;
#pragma unroll
      for (int i = 0; i < 36; ++i) { term[i] = tmp[i] * inv; E[i] += term[i]; }
    }
#pragma unroll
    for (int s = 0; s < 4; ++s) {
      mm6(E, E, tmp);
#pragma unroll
      for (int i = 0; i < 36; ++i) E[i] = tmp[i];
    }

    float* op = out + ((size_t)c * B_DIM + b0 + t) * 36;
#pragma unroll
    for (int i = 0; i < 36; ++i) op[i] = E[i];
  }
}

extern "C" void kernel_launch(void* const* d_in, const int* in_sizes, int n_in,
                              void* d_out, int out_size, void* d_ws, size_t ws_size,
                              hipStream_t stream) {
  const float* thumb = (const float*)d_in[0];
  const float* pw    = (const float*)d_in[1];
  const float* pb    = (const float*)d_in[2];
  const float* gam   = (const float*)d_in[3];
  const float* bet   = (const float*)d_in[4];
  const float* wg    = (const float*)d_in[5];
  const float* wv    = (const float*)d_in[6];
  const float* wo    = (const float*)d_in[7];
  float* out = (float*)d_out;

  // workspace layout (f16): xn | pwh | wgh | wvh | woh  (~74 MB total)
  _Float16* xn  = (_Float16*)d_ws;
  _Float16* pwh = xn  + (size_t)M_ROWS * 512;
  _Float16* wgh = pwh + (size_t)512 * 2048;
  _Float16* wvh = wgh + (size_t)4 * 512 * 512;
  _Float16* woh = wvh + (size_t)4 * 512 * 512;

  k_cvt_weights<<<dim3(1024), 256, 0, stream>>>(pw, wg, wv, wo, pwh, wgh, wvh, woh);
  k_proj_ln<<<dim3(M_ROWS / 32), 256, 0, stream>>>(thumb, pwh, pb, gam, bet, xn);
  k_swiglu_head_expm<<<dim3(B_DIM / 32, C_CAM), 256, 0, stream>>>(xn, wgh, wvh, woh, out);
}
